// AttentionLayer_22041772163392
// MI455X (gfx1250) — matrix operand order repaired and checked by reading, not compiled, not run on hardware
//
#include <hip/hip_runtime.h>
#include <hip/hip_bf16.h>
#include <math.h>

// Problem sizes (fixed by the reference)
#define B_ 32
#define T_ 4096
#define E_ 512
#define H_ 512

typedef __bf16 bf16_t;
typedef bf16_t v16bf __attribute__((ext_vector_type(16)));
typedef bf16_t v8bf  __attribute__((ext_vector_type(8)));
typedef bf16_t v4bf  __attribute__((ext_vector_type(4)));
typedef float  v8f   __attribute__((ext_vector_type(8)));

union ABfrag { v16bf v; v8bf h[2]; };

// ------------------------------------------------------------------
// K0: W [E,H] fp32  ->  Wt [H,E] bf16 (transposed, so B-fragment K
//     slices are contiguous in memory per N column)
// ------------------------------------------------------------------
__global__ void k_transpose_w(const float* __restrict__ W, bf16_t* __restrict__ Wt) {
    int gid = blockIdx.x * blockDim.x + threadIdx.x;   // over H*E
    int n = gid / E_;
    int e = gid % E_;
    Wt[gid] = (bf16_t)W[e * H_ + n];
}

// ------------------------------------------------------------------
// K1: tempU = state @ U   ([B,H] = [B,H] x [H,H]) -- tiny, scalar
// ------------------------------------------------------------------
__global__ void k_state_u(const float* __restrict__ state, const float* __restrict__ U,
                          float* __restrict__ tempU) {
    int gid = blockIdx.x * blockDim.x + threadIdx.x;   // over B*H
    int b = gid / H_;
    int h = gid % H_;
    float s = 0.f;
    for (int e = 0; e < H_; ++e) s += state[b * H_ + e] * U[e * H_ + h];
    tempU[gid] = s;
}

// ------------------------------------------------------------------
// K2: main fused kernel.
//   32-row M-block of flattened [B*T] per workgroup (2 M-tiles), so each
//   B fragment feeds 2 WMMAs. Double-buffered LDS A staging -> one
//   barrier per K-step.
//     temp = tanh(seq_tile @ W + tempU[b,:])   (bf16 WMMA, K = E)
//     xe[row] = sum_h temp[row,h] * V[h]
//   Workgroup = 8 waves; wave w covers N in [w*64, w*64+64) (4 tiles).
// ------------------------------------------------------------------
__global__ __launch_bounds__(256)
void k_scores(const float* __restrict__ seq, const bf16_t* __restrict__ Wt,
              const float* __restrict__ tempU, const float* __restrict__ V,
              float* __restrict__ xe) {
    __shared__ bf16_t lds_a[2][32 * 32];  // double-buffered A tile (bf16)
    __shared__ float  lds_red[8 * 32];    // per-wave row partial sums

    const int tid  = threadIdx.x;
    const int wave = tid >> 5;
    const int lane = tid & 31;
    const int half = lane >> 4;     // 0 or 1 (lane group)
    const int l16  = lane & 15;

    const int row0 = blockIdx.x * 32;   // base row in flattened [B*T]
    const int b    = row0 / T_;         // 32 | T so whole block is one batch
    const int nbase = wave * 64;        // this wave's N slab

    v8f c[2][4];
    for (int mt = 0; mt < 2; ++mt)
        for (int t = 0; t < 4; ++t)
            for (int i = 0; i < 8; ++i) c[mt][t][i] = 0.f;

    // tempU / V per lane: C/D layout puts N = l16 + tile offset in each lane
    float tu[4], vv[4];
    for (int t = 0; t < 4; ++t) {
        int n = nbase + t * 16 + l16;
        tu[t] = tempU[b * H_ + n];
        vv[t] = V[n];
    }

    // Staging: 32 rows x 32 K fp32 -> bf16; 1024 elems / 256 threads = 4 each.
    const int s_idx = tid * 4;
    const int s_r   = s_idx >> 5;    // row 0..31
    const int s_c   = s_idx & 31;    // col (multiple of 4)

#define STAGE_A(p, kk)                                                          \
    {                                                                           \
        const float4 f4 = *(const float4*)&seq[(size_t)(row0 + s_r) * E_ + (kk) + s_c]; \
        v4bf o;                                                                 \
        o[0] = (bf16_t)f4.x; o[1] = (bf16_t)f4.y;                               \
        o[2] = (bf16_t)f4.z; o[3] = (bf16_t)f4.w;                               \
        *(v4bf*)&lds_a[(p)][s_r * 32 + s_c] = o;                                \
        if ((kk) + 32 < E_)                                                     \
            __builtin_prefetch(&seq[(size_t)(row0 + s_r) * E_ + (kk) + 32 + s_c], 0, 1); \
    }

    STAGE_A(0, 0);
    __syncthreads();

    const int klo = half * 8;   // A per-lane K sub-offset

    int p = 0;
    for (int kk = 0; kk < E_; kk += 32, p ^= 1) {
        // Stage next K-slice into the other buffer while we compute.
        if (kk + 32 < E_) STAGE_A(p ^ 1, kk + 32);

        // A fragments (one per M-tile): lane row = l16; halves 0..7 ->
        // K = klo..klo+7, halves 8..15 -> K = 16+klo.. (ISA 16-bit A 16x32).
        ABfrag a[2];
        for (int mt = 0; mt < 2; ++mt) {
            const bf16_t* ap = &lds_a[p][(mt * 16 + l16) * 32];
            a[mt].h[0] = *(const v8bf*)(ap + klo);
            a[mt].h[1] = *(const v8bf*)(ap + 16 + klo);
        }

        // B fragments: lane holds column n = nbase+t*16+l16, halves 0..7 ->
        // K = koff..koff+7, halves 8..15 -> K = koff+16.. (matches A).
        const int koff = kk + half * 8;
        for (int t = 0; t < 4; ++t) {
            ABfrag bf;
            const bf16_t* bp = &Wt[(size_t)(nbase + t * 16 + l16) * E_ + koff];
            bf.h[0] = *(const v8bf*)(bp);
            bf.h[1] = *(const v8bf*)(bp + 16);
            for (int mt = 0; mt < 2; ++mt)
                c[mt][t] = __builtin_amdgcn_wmma_f32_16x16x32_bf16(
                    false, a[mt].v, false, bf.v, (short)0, c[mt][t], false, false);
        }
        __syncthreads();
    }
#undef STAGE_A

    // Epilogue: tanh(acc + tempU) * V, reduce over N.
    // C/D layout: VGPR r, lanes 0-15 -> M = r, lanes 16-31 -> M = r + 8.
    for (int mt = 0; mt < 2; ++mt) {
        for (int r = 0; r < 8; ++r) {
            float s = 0.f;
            for (int t = 0; t < 4; ++t)
                s += tanhf(c[mt][t][r] + tu[t]) * vv[t];
            // reduce across the 16 lanes of each half (same M row)
            for (int off = 1; off < 16; off <<= 1)
                s += __shfl_xor(s, off, 32);
            if (l16 == 0)
                lds_red[wave * 32 + mt * 16 + r + half * 8] = s;
        }
    }
    __syncthreads();
    if (tid < 32) {
        float s = 0.f;
        for (int w = 0; w < 8; ++w) s += lds_red[w * 32 + tid];
        xe[row0 + tid] = s;
    }
}

// ------------------------------------------------------------------
// K3: softmax over T per batch row
// ------------------------------------------------------------------
__global__ __launch_bounds__(256)
void k_softmax(const float* __restrict__ xe, float* __restrict__ xa) {
    __shared__ float red[256];
    const int b   = blockIdx.x;
    const int tid = threadIdx.x;
    const float* row = xe + (size_t)b * T_;

    float m = -INFINITY;
    for (int t = tid; t < T_; t += 256) m = fmaxf(m, row[t]);
    red[tid] = m; __syncthreads();
    for (int s = 128; s > 0; s >>= 1) {
        if (tid < s) red[tid] = fmaxf(red[tid], red[tid + s]);
        __syncthreads();
    }
    const float mx = red[0]; __syncthreads();

    float sum = 0.f;
    for (int t = tid; t < T_; t += 256) sum += __expf(row[t] - mx);
    red[tid] = sum; __syncthreads();
    for (int s = 128; s > 0; s >>= 1) {
        if (tid < s) red[tid] += red[tid + s];
        __syncthreads();
    }
    const float inv = 1.f / red[0];

    for (int t = tid; t < T_; t += 256)
        xa[(size_t)b * T_ + t] = __expf(row[t] - mx) * inv;
}

// ------------------------------------------------------------------
// K4: attented = seq * x_a  (pure streaming, float4)
// ------------------------------------------------------------------
__global__ void k_scale(const float* __restrict__ seq, const float* __restrict__ xa,
                        float* __restrict__ out) {
    int gid = blockIdx.x * blockDim.x + threadIdx.x;  // over B*T*E/4
    int bt  = gid / (E_ / 4);
    float a = xa[bt];
    const float4 s = ((const float4*)seq)[gid];
    float4 o;
    o.x = s.x * a; o.y = s.y * a; o.z = s.z * a; o.w = s.w * a;
    ((float4*)out)[gid] = o;
}

// ------------------------------------------------------------------
extern "C" void kernel_launch(void* const* d_in, const int* in_sizes, int n_in,
                              void* d_out, int out_size, void* d_ws, size_t ws_size,
                              hipStream_t stream) {
    const float* seq   = (const float*)d_in[0];   // [B,T,E]
    const float* state = (const float*)d_in[1];   // [B,H]
    const float* W     = (const float*)d_in[2];   // [E,H]
    const float* U     = (const float*)d_in[3];   // [H,H]
    const float* V     = (const float*)d_in[4];   // [H,1]

    char*   ws    = (char*)d_ws;
    bf16_t* Wt    = (bf16_t*)ws;                               // 512 KB
    float*  tempU = (float*)(ws + (size_t)H_ * E_ * 2);        // 64 KB
    float*  xe    = (float*)(ws + (size_t)H_ * E_ * 2 + (size_t)B_ * H_ * 4); // 512 KB

    float* xa  = (float*)d_out;                   // [B*T]
    float* att = xa + (size_t)B_ * T_;            // [B*T*E]

    k_transpose_w<<<(H_ * E_) / 256, 256, 0, stream>>>(W, Wt);
    k_state_u   <<<(B_ * H_) / 256, 256, 0, stream>>>(state, U, tempU);
    k_scores    <<<(B_ * T_) / 32, 256, 0, stream>>>(seq, Wt, tempU, V, xe);
    k_softmax   <<<B_, 256, 0, stream>>>(xe, xa);
    k_scale     <<<(B_ * T_ * (E_ / 4)) / 256, 256, 0, stream>>>(seq, xa, att);
}
